// AttLayer_28424093565642
// MI455X (gfx1250) — matrix operand order repaired and checked by reading, not compiled, not run on hardware
//
#include <hip/hip_runtime.h>
#include <hip/hip_bf16.h>

// ---------------------------------------------------------------------------
// Fused windowed-attention layer for MI455X (gfx1250, wave32, WMMA).
// One workgroup (256 thr = 8 waves) per attention block; everything staged
// through LDS (~204 KB of the 320 KB/WGP); f16 WMMA with f32 accumulation.
// ---------------------------------------------------------------------------

typedef __attribute__((ext_vector_type(16))) _Float16 v16h;
typedef __attribute__((ext_vector_type(8)))  _Float16 v8h;
typedef __attribute__((ext_vector_type(8)))  float    v8f;

#define B_   4
#define C_   64
#define CI_  32
#define L_   32768
#define BL_  128
#define NB_  256
#define W2_  256   // 2*BL

// ---- LDS layout (units: _Float16 elements; all row strides keep 16B align) --
constexpr int XQ_OFF = 0;                    // xQ  : 128 x 72  (q-side x slice)
constexpr int XW_OFF = XQ_OFF + 128 * 72;    // xW  : 256 x 72  (k/v-side window)
constexpr int WQ_OFF = XW_OFF + 256 * 72;    // Wq  : 32 x 72
constexpr int WK_OFF = WQ_OFF + 32 * 72;     // Wk  : 32 x 72
constexpr int WV_OFF = WK_OFF + 32 * 72;     // Wv  : 32 x 72
constexpr int WO_OFF = WV_OFF + 32 * 72;     // Wo  : 64 x 40
constexpr int QT_OFF = WO_OFF + 64 * 40;     // qT  : 128 x 40  [query][ch]
constexpr int KT_OFF = QT_OFF + 128 * 40;    // kT  : 256 x 40  [kpos][ch]
constexpr int VC_OFF = KT_OFF + 256 * 40;    // vch : 32 x 272  [ch][kpos]
constexpr int AT_OFF = VC_OFF + 32 * 272;    // aT  : 128 x 272 [query][kpos]
constexpr int OT_OFF = AT_OFF + 128 * 272;   // outT: 128 x 40  [query][ch]
constexpr int H_TOTAL = OT_OFF + 128 * 40;   // = 101120 halves (16B aligned)

constexpr int F_BQ = 0, F_BK = 32, F_BV = 64, F_BO = 96, F_MW = 160, F_MO = 416;
constexpr int F_TOTAL = 544;
constexpr size_t SMEM_BYTES = (size_t)H_TOTAL * 2 + (size_t)F_TOTAL * 4; // 204416

// ---- WMMA helper (codegen-confirmed builtin, 8-arg form) -------------------
__device__ __forceinline__ v8f wmma_f16(v16h a, v16h b, v8f c) {
  return __builtin_amdgcn_wmma_f32_16x16x32_f16(
      /*neg_a=*/false, a, /*neg_b=*/false, b,
      /*c_mod=*/(short)0, c, /*reuse_a=*/false, /*reuse_b=*/false);
}

// A fragment (16x32, M x K). ISA layout: lane l holds row m0 + (l&15);
// lanes 0-15: K in {k0..k0+7} U {k0+16..k0+23}; lanes 16-31: shifted by +8.
__device__ __forceinline__ v16h load_frag_a(const _Float16* base, int rowStride,
                                            int m0, int k0, int lane) {
  const _Float16* row = base + (m0 + (lane & 15)) * rowStride;
  int c1 = k0 + ((lane >> 4) << 3);
  int c2 = c1 + 16;
  v16h f;
#pragma unroll
  for (int i = 0; i < 8; ++i) { f[i] = row[c1 + i]; f[i + 8] = row[c2 + i]; }
  return f;
}

// B fragment (32x16, K x N), stored as [n][k] rows. Lane l holds col n0+(l&15),
// lanes 0-15: K in {k0..k0+7} U {k0+16..k0+23}; lanes 16-31: shifted by +8.
__device__ __forceinline__ v16h load_frag_b(const _Float16* base, int rowStride,
                                            int k0, int n0, int lane) {
  const _Float16* row = base + (n0 + (lane & 15)) * rowStride;
  int c1 = k0 + ((lane >> 4) << 3);
  int c2 = c1 + 16;
  v16h f;
#pragma unroll
  for (int i = 0; i < 8; ++i) { f[i] = row[c1 + i]; f[i + 8] = row[c2 + i]; }
  return f;
}

__global__ __launch_bounds__(256) void fused_window_attn(
    const float* __restrict__ x1, const float* __restrict__ mask,
    const float* __restrict__ Wq, const float* __restrict__ bq,
    const float* __restrict__ Wk, const float* __restrict__ bk,
    const float* __restrict__ Wv, const float* __restrict__ bv,
    const float* __restrict__ Wo, const float* __restrict__ bo,
    float* __restrict__ out) {
  extern __shared__ __align__(16) char smem[];
  _Float16* hb = (_Float16*)smem;
  float* fb = (float*)(smem + (size_t)H_TOTAL * 2);

  _Float16* xQ   = hb + XQ_OFF;
  _Float16* xW   = hb + XW_OFF;
  _Float16* wqs  = hb + WQ_OFF;
  _Float16* wks  = hb + WK_OFF;
  _Float16* wvs  = hb + WV_OFF;
  _Float16* wos  = hb + WO_OFF;
  _Float16* qT   = hb + QT_OFF;
  _Float16* kT   = hb + KT_OFF;
  _Float16* vch  = hb + VC_OFF;
  _Float16* aT   = hb + AT_OFF;
  _Float16* outT = hb + OT_OFF;
  float* bqs   = fb + F_BQ;
  float* bks   = fb + F_BK;
  float* bvs   = fb + F_BV;
  float* bos   = fb + F_BO;
  float* maskw = fb + F_MW;   // k-side window mask (256)
  float* masko = fb + F_MO;   // q-side output mask (128)

  const int tid  = threadIdx.x;
  const int wave = __builtin_amdgcn_readfirstlane(tid >> 5);
  const int lane = tid & 31;

  // Block pairing replicates the reference's (B,nb) vs (nb,B) reshape quirk:
  // n = b*nb + blk;  k/v window comes from (blk_k = n/B, b_k = n%B).
  const int b    = blockIdx.x >> 8;
  const int blk  = blockIdx.x & 255;
  const int bkb  = blk & 3;                 // k/v batch
  const int blkk = (b << 6) + (blk >> 2);   // k/v block
  const int baseK = blkk * BL_ - (BL_ / 2);
  const int baseQ = blk * BL_;

  const float* xb_q = x1 + (size_t)b   * C_ * L_;
  const float* xb_k = x1 + (size_t)bkb * C_ * L_;

  // ---- Stage 0: stage x-window, weights, biases, masks into LDS (fp32->f16)
  // x window: 4 bounds-checked scalar loads per thread along position axis (zero-padded).
  for (int idx = tid; idx < C_ * (W2_ / 4); idx += 256) {
    int c = idx >> 6, j4 = (idx & 63) << 2;   // 64 quads per channel row
    int p = baseK + j4;
#pragma unroll
    for (int u = 0; u < 4; ++u) {
      int pp = p + u;
      float v = ((unsigned)pp < (unsigned)L_) ? xb_k[(size_t)c * L_ + pp] : 0.0f;
      xW[(j4 + u) * 72 + c] = (_Float16)v;
    }
  }
  for (int idx = tid; idx < C_ * (BL_ / 4); idx += 256) {
    int c = idx >> 5, j4 = (idx & 31) << 2;
    const float* src = xb_q + (size_t)c * L_ + baseQ + j4;
    float4 v = *(const float4*)src;           // always in range
    xQ[(j4 + 0) * 72 + c] = (_Float16)v.x;
    xQ[(j4 + 1) * 72 + c] = (_Float16)v.y;
    xQ[(j4 + 2) * 72 + c] = (_Float16)v.z;
    xQ[(j4 + 3) * 72 + c] = (_Float16)v.w;
  }
  for (int idx = tid; idx < CI_ * C_; idx += 256) {
    int o = idx >> 6, c = idx & 63;
    wqs[o * 72 + c] = (_Float16)Wq[idx];
    wks[o * 72 + c] = (_Float16)Wk[idx];
    wvs[o * 72 + c] = (_Float16)Wv[idx];
    wos[(idx >> 5) * 40 + (idx & 31)] = (_Float16)Wo[idx];  // Wo is (64,32)
  }
  if (tid < CI_) { bqs[tid] = bq[tid]; bks[tid] = bk[tid]; bvs[tid] = bv[tid]; }
  if (tid < C_) bos[tid] = bo[tid];
  {
    int p = baseK + tid;
    maskw[tid] = ((unsigned)p < (unsigned)L_) ? mask[(size_t)bkb * L_ + p] : 0.0f;
  }
  if (tid < BL_) masko[tid] = mask[(size_t)b * L_ + baseQ + tid];
  __syncthreads();

  const int rh = (lane >> 4) << 3;  // row offset within D tile for this half

  // ---- Stage 1: QKV projections. 80 WMMA tile-jobs over 8 waves.
  for (int j0 = 0; j0 < 80; j0 += 8) {
    int job = j0 + wave;                                   // wave-uniform
    int which, mt, nt;
    if (job < 16)      { which = 0; mt = job >> 3;        nt = job & 7;  }
    else if (job < 48) { which = 1; mt = (job - 16) >> 4; nt = (job - 16) & 15; }
    else               { which = 2; mt = (job - 48) >> 4; nt = (job - 48) & 15; }
    which = __builtin_amdgcn_readfirstlane(which);
    mt    = __builtin_amdgcn_readfirstlane(mt);
    nt    = __builtin_amdgcn_readfirstlane(nt);

    const _Float16* Wm = (which == 0) ? wqs : ((which == 1) ? wks : wvs);
    const _Float16* Xs = (which == 0) ? xQ : xW;
    const float* bias  = (which == 0) ? bqs : ((which == 1) ? bks : bvs);

    v8f acc = {};
    v16h a0 = load_frag_a(Wm, 72, mt * 16, 0, lane);
    v16h b0 = load_frag_b(Xs, 72, 0, nt * 16, lane);
    acc = wmma_f16(a0, b0, acc);
    v16h a1 = load_frag_a(Wm, 72, mt * 16, 32, lane);
    v16h b1 = load_frag_b(Xs, 72, 32, nt * 16, lane);
    acc = wmma_f16(a1, b1, acc);

    int col = nt * 16 + (lane & 15);
    if (which <= 1) {
      // q/k target layout [pos][ch]: 8 rows are contiguous -> one b128 store.
      _Float16* dst = ((which == 0) ? qT : kT) + col * 40 + mt * 16 + rh;
      v8h h;
#pragma unroll
      for (int r = 0; r < 8; ++r)
        h[r] = (_Float16)(acc[r] + bias[mt * 16 + rh + r]);
      *(v8h*)dst = h;
    } else {
      // v target layout [ch][kpos]: stride 272 per row -> scalar stores.
#pragma unroll
      for (int r = 0; r < 8; ++r) {
        int m = mt * 16 + rh + r;
        vch[m * 272 + col] = (_Float16)(acc[r] + bias[m]);
      }
    }
  }
  __syncthreads();

  // ---- Stage 2: energy + masked softmax. Wave w owns query tile qt = w.
  const int qt = wave;
  const float scale = 0.17677669529663687f;  // 1/sqrt(32)
  {
    v16h aq = load_frag_a(qT, 40, qt * 16, 0, lane);
    v8f e[16];
#pragma unroll
    for (int nt = 0; nt < 16; ++nt) {
      v16h bk_ = load_frag_b(kT, 40, 0, nt * 16, lane);
      v8f z = {};
      e[nt] = wmma_f16(aq, bk_, z);
    }
    // scale + log(window_mask + 1e-6), in place
#pragma unroll
    for (int nt = 0; nt < 16; ++nt) {
      int kj = nt * 16 + (lane & 15);
      float pm = maskw[kj];
#pragma unroll
      for (int r = 0; r < 8; ++r) {
        int qi = qt * 16 + rh + r;
        float fm = (kj >= qi && kj < qi + BL_) ? pm : 0.0f;
        e[nt][r] = e[nt][r] * scale + __logf(fm + 1e-6f);
      }
    }
    // per-row softmax; row lives across 16 lanes of one half + 16 N-tiles
#pragma unroll
    for (int r = 0; r < 8; ++r) {
      float mx = -3.0e38f;
#pragma unroll
      for (int nt = 0; nt < 16; ++nt) mx = fmaxf(mx, e[nt][r]);
      mx = fmaxf(mx, __shfl_xor(mx, 1, 32));
      mx = fmaxf(mx, __shfl_xor(mx, 2, 32));
      mx = fmaxf(mx, __shfl_xor(mx, 4, 32));
      mx = fmaxf(mx, __shfl_xor(mx, 8, 32));
      float s = 0.0f;
#pragma unroll
      for (int nt = 0; nt < 16; ++nt) {
        float p = __expf(e[nt][r] - mx);
        e[nt][r] = p;
        s += p;
      }
      s += __shfl_xor(s, 1, 32);
      s += __shfl_xor(s, 2, 32);
      s += __shfl_xor(s, 4, 32);
      s += __shfl_xor(s, 8, 32);
      float inv = 1.0f / s;
      int qi = qt * 16 + rh + r;
#pragma unroll
      for (int nt = 0; nt < 16; ++nt) {
        int kj = nt * 16 + (lane & 15);
        float pm = maskw[kj];
        float fm = (kj >= qi && kj < qi + BL_) ? pm : 0.0f;
        aT[qi * 272 + kj] = (_Float16)(e[nt][r] * inv * fm);  // [query][kpos]
      }
    }
  }
  __syncthreads();

  // ---- Stage 3: out = V (32x256) @ attn^T (256 x 16 queries), relu -> LDS
#pragma unroll
  for (int mt = 0; mt < 2; ++mt) {
    v8f acc = {};
#pragma unroll
    for (int ks = 0; ks < 8; ++ks) {
      v16h av = load_frag_a(vch, 272, mt * 16, ks * 32, lane);
      v16h bt = load_frag_b(aT, 272, ks * 32, qt * 16, lane);
      acc = wmma_f16(av, bt, acc);
    }
    // outT layout [query][ch]: 8 rows contiguous -> one b128 store.
    int q = qt * 16 + (lane & 15);
    v8h h;
#pragma unroll
    for (int r = 0; r < 8; ++r) h[r] = (_Float16)fmaxf(acc[r], 0.0f);
    *(v8h*)(outT + q * 40 + mt * 16 + rh) = h;
  }
  __syncthreads();

  // ---- Stage 4: y = Wo (64x32) @ relu(out) + bo, masked, store fp32
  {
    v16h bfrag = load_frag_b(outT, 40, 0, qt * 16, lane);
    int q = qt * 16 + (lane & 15);
    float mval = masko[q];
    float* outp = out + (size_t)b * C_ * L_ + (size_t)blk * BL_;
#pragma unroll
    for (int mt = 0; mt < 4; ++mt) {
      v16h aw = load_frag_a(wos, 40, mt * 16, 0, lane);
      v8f z = {};
      v8f y = wmma_f16(aw, bfrag, z);
#pragma unroll
      for (int r = 0; r < 8; ++r) {
        int o = mt * 16 + rh + r;
        outp[(size_t)o * L_ + q] = (y[r] + bos[o]) * mval;
      }
    }
  }
}

extern "C" void kernel_launch(void* const* d_in, const int* in_sizes, int n_in,
                              void* d_out, int out_size, void* d_ws, size_t ws_size,
                              hipStream_t stream) {
  (void)in_sizes; (void)n_in; (void)out_size; (void)d_ws; (void)ws_size;
  const float* x1   = (const float*)d_in[0];
  const float* mask = (const float*)d_in[1];
  const float* Wq   = (const float*)d_in[2];
  const float* bq   = (const float*)d_in[3];
  const float* Wk   = (const float*)d_in[4];
  const float* bk   = (const float*)d_in[5];
  const float* Wv   = (const float*)d_in[6];
  const float* bv   = (const float*)d_in[7];
  const float* Wo   = (const float*)d_in[8];
  const float* bo   = (const float*)d_in[9];
  float* out = (float*)d_out;

  // Allow >64KB dynamic LDS (gfx1250 WGP has 320KB).
  (void)hipFuncSetAttribute((const void*)fused_window_attn,
                            hipFuncAttributeMaxDynamicSharedMemorySize,
                            (int)SMEM_BYTES);

  dim3 grid(B_ * NB_);   // 1024 blocks: one per (b, blk)
  dim3 block(256);       // 8 waves of 32
  fused_window_attn<<<grid, block, SMEM_BYTES, stream>>>(
      x1, mask, Wq, bq, Wk, bk, Wv, bv, Wo, bo, out);
}
